// HSI_MambaVisionBlock_62663572848659
// MI455X (gfx1250) — matrix operand order repaired and checked by reading, not compiled, not run on hardware
//
#include <hip/hip_runtime.h>
#include <hip/hip_bf16.h>
#include <math.h>

typedef __attribute__((ext_vector_type(16))) _Float16 v16h;
typedef __attribute__((ext_vector_type(8)))  _Float16 v8h;
typedef __attribute__((ext_vector_type(8)))  float    v8f;

#define M_ROWS   32768      // B*L = 8*4096
#define D_MODEL  256
#define D_FF     1024
#define SEQ_L    4096
#define NBATCH   8
#define NCHUNK   64         // scan chunks per sequence
#define CHLEN    64         // elements per chunk (NCHUNK*CHLEN == SEQ_L)

// ---------------- workspace layout (bytes) ----------------
#define OFF_WIN   0u            // in_proj_w  f16 [512,256]   262144
#define OFF_WOUT  262144u       // out_proj_w f16 [256,256]   131072
#define OFF_WFC1  393216u       // fc1_w      f16 [1024,256]  524288
#define OFF_WFC2  917504u       // fc2_w      f16 [256,1024]  524288
#define OFF_H     1572864u      // h  f16 [32768,256]   16 MB (ln1 & ln2 out)
#define OFF_XZ    18350080u     // xz f16 [32768,512]   32 MB
#define OFF_YZ    51904512u     // yz f16 [32768,256]   16 MB
#define OFF_X1    68681728u     // x1 f32 [32768,256]   32 MB (post-mixer residual)
#define OFF_G     102236160u    // g  f16 [32768,1024]  64 MB (gelu output)
#define OFF_CS    169345024u    // chunk sums f32 [8,64,256]  512 KB

__device__ __forceinline__ float silu_f(float v) { return v / (1.0f + __expf(-v)); }

// ---------------- f32 -> f16 weight conversion ----------------
__global__ void cvt_f32_to_f16(const float* __restrict__ src,
                               _Float16* __restrict__ dst, int n) {
    int i = blockIdx.x * blockDim.x + threadIdx.x;
    if (i < n) dst[i] = (_Float16)src[i];
}

// ---------------- LayerNorm (one row of 256 per block) ----------------
__global__ __launch_bounds__(256)
void ln_to_f16(const float* __restrict__ x, const float* __restrict__ w,
               const float* __restrict__ b, _Float16* __restrict__ out) {
    const int row = blockIdx.x;
    const int tid = threadIdx.x;
    float v = x[(size_t)row * D_MODEL + tid];
    float s = v, s2 = v * v;
    #pragma unroll
    for (int o = 16; o > 0; o >>= 1) {
        s  += __shfl_xor(s,  o, 32);
        s2 += __shfl_xor(s2, o, 32);
    }
    __shared__ float ss[8], ss2[8];
    if ((tid & 31) == 0) { ss[tid >> 5] = s; ss2[tid >> 5] = s2; }
    __syncthreads();
    float tot = 0.f, tot2 = 0.f;
    #pragma unroll
    for (int i = 0; i < 8; i++) { tot += ss[i]; tot2 += ss2[i]; }
    float mu  = tot  * (1.0f / D_MODEL);
    float var = tot2 * (1.0f / D_MODEL) - mu * mu;
    float rs  = rsqrtf(var + 1e-5f);
    out[(size_t)row * D_MODEL + tid] = (_Float16)((v - mu) * rs * w[tid] + b[tid]);
}

// ---------------- WMMA GEMM: out[M,N] = A[M,K] * W[N,K]^T (+epilogue) ----
// B tile (64 x K-chunk) staged in LDS once per block & chunk, shared by 8 waves.
// A fragments (global) AND B fragments (LDS) double-buffered in registers so
// each iteration's loads overlap the previous iteration's 4 WMMAs.
// MODE 0: store f16   MODE 1: +resid f32   MODE 2: +bias GELU f16   MODE 3: +bias+resid f32
#define KCH   256
#define LPAD  8                     // +8 halves/row => 4-bank shift, conflict-free
#define LROW  (KCH + LPAD)          // 264 halves = 528 B

template<int KDIM, int NDIM, int MODE>
__global__ __launch_bounds__(256)
void gemm_wmma(const _Float16* __restrict__ A,
               const _Float16* __restrict__ W,
               const float* __restrict__ bias,
               const float* __restrict__ resid,
               void* __restrict__ outp) {
    __shared__ _Float16 wtile[64 * LROW];

    const int tid  = threadIdx.x;
    const int lane = tid & 31;
    const int wave = tid >> 5;
    const int m0   = (blockIdx.x * 8 + wave) * 16;
    const int n0   = blockIdx.y * 64;
    const int r    = lane & 15;
    const int hi   = lane >> 4;

    v8f acc[4] = {};
    const _Float16* arow = A + (size_t)(m0 + r) * KDIM;

    for (int kc = 0; kc < KDIM; kc += KCH) {
        // ---- cooperative stage: W[n0:n0+64, kc:kc+256] -> LDS (padded rows)
        __syncthreads();
        #pragma unroll
        for (int it = 0; it < 8; it++) {          // 64 rows * 32 segs / 256 thr
            int idx = tid + it * 256;
            int row = idx >> 5;
            int seg = idx & 31;
            *(v8h*)&wtile[row * LROW + seg * 8] =
                *(const v8h*)(W + (size_t)(n0 + row) * KDIM + kc + seg * 8);
        }
        __syncthreads();

        // ---- prologue: A and B fragments for kk=0
        v16h a_cur;
        {
            v8h alo = *(const v8h*)(arow + kc + hi * 8);
            v8h ahi = *(const v8h*)(arow + kc + 16 + hi * 8);
            #pragma unroll
            for (int j = 0; j < 8; j++) { a_cur[j] = alo[j]; a_cur[8 + j] = ahi[j]; }
        }
        v16h b_cur[4];
        #pragma unroll
        for (int nn = 0; nn < 4; nn++) {
            const _Float16* lrow = &wtile[(nn * 16 + r) * LROW + hi * 8];
            v8h blo = *(const v8h*)lrow;
            v8h bhi = *(const v8h*)(lrow + 16);
            #pragma unroll
            for (int j = 0; j < 8; j++) { b_cur[nn][j] = blo[j]; b_cur[nn][8 + j] = bhi[j]; }
        }

        #pragma unroll
        for (int kk = 0; kk < KCH; kk += 32) {
            const bool last = (kk + 32 >= KCH);
            v16h a_next;
            v16h b_next[4];
            if (!last) {                           // prefetch next A (global)
                v8h nlo = *(const v8h*)(arow + kc + kk + 32 + hi * 8);
                v8h nhi = *(const v8h*)(arow + kc + kk + 48 + hi * 8);
                #pragma unroll
                for (int j = 0; j < 8; j++) { a_next[j] = nlo[j]; a_next[8 + j] = nhi[j]; }
                #pragma unroll
                for (int nn = 0; nn < 4; nn++) {   // prefetch next B (LDS)
                    const _Float16* lrow =
                        &wtile[(nn * 16 + r) * LROW + kk + 32 + hi * 8];
                    v8h blo = *(const v8h*)lrow;
                    v8h bhi = *(const v8h*)(lrow + 16);
                    #pragma unroll
                    for (int j = 0; j < 8; j++) {
                        b_next[nn][j] = blo[j]; b_next[nn][8 + j] = bhi[j];
                    }
                }
            }
            #pragma unroll
            for (int nn = 0; nn < 4; nn++) {
                acc[nn] = __builtin_amdgcn_wmma_f32_16x16x32_f16(
                    false, a_cur, false, b_cur[nn], (short)0, acc[nn], false, false);
            }
            if (!last) {
                a_cur = a_next;
                #pragma unroll
                for (int nn = 0; nn < 4; nn++) b_cur[nn] = b_next[nn];
            }
        }
    }

    #pragma unroll
    for (int nn = 0; nn < 4; nn++) {
        int col = n0 + nn * 16 + r;
        #pragma unroll
        for (int j = 0; j < 8; j++) {
            int row = m0 + hi * 8 + j;
            float v = acc[nn][j];
            size_t idx = (size_t)row * NDIM + col;
            if (MODE == 0) {
                ((_Float16*)outp)[idx] = (_Float16)v;
            } else if (MODE == 1) {
                ((float*)outp)[idx] = v + resid[idx];
            } else if (MODE == 2) {
                float t = v + bias[col];
                float g = 0.5f * t * (1.0f + erff(t * 0.70710678118654752f));
                ((_Float16*)outp)[idx] = (_Float16)g;
            } else {
                ((float*)outp)[idx] = v + bias[col] + resid[idx];
            }
        }
    }
}

// ---------------- 3-phase chunked scan (conv + SiLU + cumsum + gate) -----
// phase 1: per (b, chunk, d) partial sums of silu(conv(xc))
__global__ __launch_bounds__(256)
void scan_chunksum(const _Float16* __restrict__ xz,
                   const float* __restrict__ conv_w, const float* __restrict__ conv_b,
                   float* __restrict__ csum) {
    int t = blockIdx.x * 256 + threadIdx.x;   // < NBATCH*NCHUNK*D_MODEL
    int d = t & 255;
    int c = (t >> 8) & (NCHUNK - 1);
    int b = t >> 14;
    float w0 = conv_w[d * 3 + 0], w1 = conv_w[d * 3 + 1], w2 = conv_w[d * 3 + 2];
    float cb = conv_b[d];
    const _Float16* base = xz + (size_t)b * SEQ_L * (2 * D_MODEL);
    int l0 = c * CHLEN;
    float xm1 = (l0 >= 1) ? (float)base[(size_t)(l0 - 1) * 512 + d] : 0.f;
    float xm2 = (l0 >= 2) ? (float)base[(size_t)(l0 - 2) * 512 + d] : 0.f;
    float s = 0.f;
    for (int l = l0; l < l0 + CHLEN; l++) {
        float xc = (float)base[(size_t)l * 512 + d];
        float cv = w0 * xm2 + w1 * xm1 + w2 * xc + cb;
        xm2 = xm1; xm1 = xc;
        s += silu_f(cv);
    }
    csum[((size_t)b * NCHUNK + c) * D_MODEL + d] = s;
}

// phase 2: exclusive prefix over chunk sums, in place (one thread per (b,d))
__global__ __launch_bounds__(256)
void scan_prefix(float* __restrict__ csum) {
    int t = blockIdx.x * 256 + threadIdx.x;   // < NBATCH*D_MODEL
    int b = t >> 8, d = t & 255;
    float run = 0.f;
    for (int c = 0; c < NCHUNK; c++) {
        size_t i = ((size_t)b * NCHUNK + c) * D_MODEL + d;
        float v = csum[i];
        csum[i] = run;
        run += v;
    }
}

// phase 3: recompute conv+silu seeded with prefix, emit gated yz (f16)
__global__ __launch_bounds__(256)
void scan_emit(const _Float16* __restrict__ xz,
               const float* __restrict__ conv_w, const float* __restrict__ conv_b,
               const float* __restrict__ ssmB,  const float* __restrict__ ssmC,
               const float* __restrict__ ssmD,  const float* __restrict__ csum,
               _Float16* __restrict__ yz) {
    int t = blockIdx.x * 256 + threadIdx.x;
    int d = t & 255;
    int c = (t >> 8) & (NCHUNK - 1);
    int b = t >> 14;
    float w0 = conv_w[d * 3 + 0], w1 = conv_w[d * 3 + 1], w2 = conv_w[d * 3 + 2];
    float cb = conv_b[d];
    float bc = 0.f;
    #pragma unroll
    for (int s = 0; s < 8; s++) bc += ssmB[d * 8 + s] * ssmC[d * 8 + s];
    float Dd = ssmD[d];
    const _Float16* base = xz + (size_t)b * SEQ_L * (2 * D_MODEL);
    _Float16* ob = yz + (size_t)b * SEQ_L * D_MODEL;
    int l0 = c * CHLEN;
    float xm1 = (l0 >= 1) ? (float)base[(size_t)(l0 - 1) * 512 + d] : 0.f;
    float xm2 = (l0 >= 2) ? (float)base[(size_t)(l0 - 2) * 512 + d] : 0.f;
    float cum = csum[((size_t)b * NCHUNK + c) * D_MODEL + d];
    for (int l = l0; l < l0 + CHLEN; l++) {
        const _Float16* p = base + (size_t)l * 512;
        float xc = (float)p[d];
        float zz = (float)p[D_MODEL + d];
        float cv = w0 * xm2 + w1 * xm1 + w2 * xc + cb;
        xm2 = xm1; xm1 = xc;
        float a = silu_f(cv);
        cum += a;                                  // inclusive cumsum
        float y = cum * bc + a * Dd;
        ob[(size_t)l * D_MODEL + d] = (_Float16)(y * silu_f(zz));
    }
}

// ---------------------------------------------------------------------------
extern "C" void kernel_launch(void* const* d_in, const int* in_sizes, int n_in,
                              void* d_out, int out_size, void* d_ws, size_t ws_size,
                              hipStream_t stream) {
    const float* x         = (const float*)d_in[0];
    const float* ln1_w     = (const float*)d_in[1];
    const float* ln1_b     = (const float*)d_in[2];
    const float* in_proj_w = (const float*)d_in[3];
    const float* conv_w    = (const float*)d_in[4];
    const float* conv_b    = (const float*)d_in[5];
    const float* ssm_B     = (const float*)d_in[6];
    const float* ssm_C     = (const float*)d_in[7];
    const float* ssm_D     = (const float*)d_in[8];
    const float* out_proj_w= (const float*)d_in[9];
    const float* ln2_w     = (const float*)d_in[10];
    const float* ln2_b     = (const float*)d_in[11];
    const float* fc1_w     = (const float*)d_in[12];
    const float* fc1_b     = (const float*)d_in[13];
    const float* fc2_w     = (const float*)d_in[14];
    const float* fc2_b     = (const float*)d_in[15];

    char* ws = (char*)d_ws;
    _Float16* w16_in  = (_Float16*)(ws + OFF_WIN);
    _Float16* w16_out = (_Float16*)(ws + OFF_WOUT);
    _Float16* w16_fc1 = (_Float16*)(ws + OFF_WFC1);
    _Float16* w16_fc2 = (_Float16*)(ws + OFF_WFC2);
    _Float16* bufH    = (_Float16*)(ws + OFF_H);
    _Float16* bufXZ   = (_Float16*)(ws + OFF_XZ);
    _Float16* bufYZ   = (_Float16*)(ws + OFF_YZ);
    float*    bufX1   = (float*)   (ws + OFF_X1);
    _Float16* bufG    = (_Float16*)(ws + OFF_G);
    float*    bufCS   = (float*)   (ws + OFF_CS);

    // 0) weights -> f16 (row-major [N,K] matches WMMA B fragment layout)
    cvt_f32_to_f16<<<(512*256 + 255)/256, 256, 0, stream>>>(in_proj_w,  w16_in,  512*256);
    cvt_f32_to_f16<<<(256*256 + 255)/256, 256, 0, stream>>>(out_proj_w, w16_out, 256*256);
    cvt_f32_to_f16<<<(1024*256+ 255)/256, 256, 0, stream>>>(fc1_w,      w16_fc1, 1024*256);
    cvt_f32_to_f16<<<(256*1024+ 255)/256, 256, 0, stream>>>(fc2_w,      w16_fc2, 256*1024);

    // 1) h = LN1(x) -> f16
    ln_to_f16<<<M_ROWS, 256, 0, stream>>>(x, ln1_w, ln1_b, bufH);

    // 2) xz = h @ in_proj_w^T        [32768,512] f16
    gemm_wmma<256, 512, 0><<<dim3(M_ROWS/128, 512/64), 256, 0, stream>>>(
        bufH, w16_in, nullptr, nullptr, bufXZ);

    // 3) chunked scan: conv+silu+cumsum+gate -> yz f16
    {
        int nthr = NBATCH * NCHUNK * D_MODEL;     // 131072
        scan_chunksum<<<nthr/256, 256, 0, stream>>>(bufXZ, conv_w, conv_b, bufCS);
        scan_prefix<<<(NBATCH*D_MODEL)/256, 256, 0, stream>>>(bufCS);
        scan_emit<<<nthr/256, 256, 0, stream>>>(bufXZ, conv_w, conv_b,
                                                ssm_B, ssm_C, ssm_D, bufCS, bufYZ);
    }

    // 4) x1 = x + yz @ out_proj_w^T  [32768,256] f32
    gemm_wmma<256, 256, 1><<<dim3(M_ROWS/128, 256/64), 256, 0, stream>>>(
        bufYZ, w16_out, nullptr, x, bufX1);

    // 5) h2 = LN2(x1) -> f16
    ln_to_f16<<<M_ROWS, 256, 0, stream>>>(bufX1, ln2_w, ln2_b, bufH);

    // 6) g = gelu(h2 @ fc1_w^T + fc1_b)  [32768,1024] f16
    gemm_wmma<256, 1024, 2><<<dim3(M_ROWS/128, 1024/64), 256, 0, stream>>>(
        bufH, w16_fc1, fc1_b, nullptr, bufG);

    // 7) out = x1 + g @ fc2_w^T + fc2_b  [32768,256] f32
    gemm_wmma<1024, 256, 3><<<dim3(M_ROWS/128, 256/64), 256, 0, stream>>>(
        bufG, w16_fc2, fc2_b, bufX1, (float*)d_out);
}
